// TransformerAttention_37520834298572
// MI455X (gfx1250) — compile-verified, matrix operand order checked by reading, not run on hardware
//
#include <hip/hip_runtime.h>
#include <hip/hip_bf16.h>
#include <stdint.h>

// Problem constants (match reference)
#define B_   2
#define T_   2048
#define D_   2048
#define H_   16
#define KV_  4
#define DH_  128
#define REP_ (H_ / KV_)
#define NQ_  (H_ * DH_)    // 2048
#define NKV_ (KV_ * DH_)   // 512
#define MTOK_ (B_ * T_)    // 4096

typedef __attribute__((ext_vector_type(16))) __bf16 v16bf;
typedef __attribute__((ext_vector_type(8)))  float  v8f;

union FragB { v16bf v; unsigned int u[8]; };
union FragC { v8f   v; float f[8]; };

static __device__ __forceinline__ unsigned short f2bf(float f) {
  union { float f; unsigned int u; } a; a.f = f;
  unsigned int u = a.u;
  unsigned int r = (u + 0x7FFFu + ((u >> 16) & 1u)) >> 16;   // RNE
  return (unsigned short)r;
}

// Load a 16x32 bf16 fragment (A-layout; B uses the same mapping with row=n).
// gfx1250 16-bit matrix layout: lanes 0-15 row=lane, K 0..7 (v0-3) & 16..23 (v4-7);
// lanes 16-31 same rows, K 8..15 & 24..31. Pairs are contiguous dwords in memory,
// so this lowers to two global_load_b128 per fragment.
static __device__ __forceinline__ v16bf load_frag_16x32(const unsigned short* base, int ld) {
  const int lane = threadIdx.x & 31;
  const int row  = lane & 15;
  const int half = lane >> 4;
  const unsigned int* p = (const unsigned int*)(base + (size_t)row * ld);
  FragB f;
#pragma unroll
  for (int r = 0; r < 8; ++r) {
    const int k = ((r & 4) ? 16 : 0) + half * 8 + (r & 3) * 2;
    f.u[r] = p[k >> 1];
  }
  return f.v;
}

// ---------------- elementwise conversion kernels ----------------

__global__ void cvt_bf16_kernel(const float* __restrict__ in,
                                unsigned short* __restrict__ out, int n) {
  int i = blockIdx.x * blockDim.x + threadIdx.x;
  if (i < n) out[i] = f2bf(in[i]);
}

// W [K,N] f32 -> Wt [N,K] bf16 (transpose + convert)
__global__ void wt_cvt_kernel(const float* __restrict__ W,
                              unsigned short* __restrict__ Wt, int K, int N) {
  int i = blockIdx.x * blockDim.x + threadIdx.x;
  if (i >= K * N) return;
  int n = i % N, k = i / N;
  Wt[(size_t)n * K + k] = f2bf(W[i]);
}

// RoPE + convert: src f32 [B*T, heads*Dh] -> dst bf16 [B, heads, T, Dh]
__global__ void rope_cvt_kernel(const float* __restrict__ src,
                                unsigned short* __restrict__ dst, int heads) {
  int i = blockIdx.x * blockDim.x + threadIdx.x;
  int total = B_ * T_ * heads * (DH_ / 2);
  if (i >= total) return;
  int j  = i % (DH_ / 2);
  int r  = i / (DH_ / 2);
  int hh = r % heads;
  int t  = (r / heads) % T_;
  int b  = r / (heads * T_);
  float inv = __powf(10000.0f, -(float)(2 * j) / (float)DH_);
  float ang = (float)t * inv;
  float s, c;
  __sincosf(ang, &s, &c);
  const float* p = src + (((size_t)b * T_ + t) * heads + hh) * DH_ + 2 * j;
  float x1 = p[0], x2 = p[1];
  unsigned short* q = dst + (((size_t)b * heads + hh) * T_ + t) * DH_ + 2 * j;
  q[0] = f2bf(x1 * c - x2 * s);
  q[1] = f2bf(x1 * s + x2 * c);
}

// V f32 [B*T, KV*Dh] -> Vt bf16 [B, KV, Dh, T]
__global__ void vtrans_cvt_kernel(const float* __restrict__ v,
                                  unsigned short* __restrict__ vt) {
  int i = blockIdx.x * blockDim.x + threadIdx.x;
  if (i >= MTOK_ * NKV_) return;
  int dh = i % DH_;
  int r  = i / DH_;
  int kv = r % KV_;
  int t  = (r / KV_) % T_;
  int b  = r / (KV_ * T_);
  vt[(((size_t)b * KV_ + kv) * DH_ + dh) * T_ + t] = f2bf(v[i]);
}

// ---------------- generic bf16 WMMA GEMM (ping-pong pipelined) ----------------
// C[M,N](f32) = A[M,K](bf16,row-major) x Bt[N,K](bf16,row-major == B^T)
// 8 waves/block; each wave: 32(M) x 64(N) tile = 8 accumulators.
// k unrolled by 2 with two independent fragment sets (no register rotation
// copies); each mma_set only waits on loads issued a half-iteration earlier.
// Requires K % 64 == 0 and K >= 128 (true for all uses here: K = 2048).

struct FragSet { v16bf a0, a1, b0, b1, b2, b3; };

static __device__ __forceinline__ void load_set(FragSet& s,
    const unsigned short* A0, const unsigned short* A1,
    const unsigned short* B0, const unsigned short* B1,
    const unsigned short* B2, const unsigned short* B3,
    int k, int K) {
  s.a0 = load_frag_16x32(A0 + k, K);
  s.a1 = load_frag_16x32(A1 + k, K);
  s.b0 = load_frag_16x32(B0 + k, K);
  s.b1 = load_frag_16x32(B1 + k, K);
  s.b2 = load_frag_16x32(B2 + k, K);
  s.b3 = load_frag_16x32(B3 + k, K);
}

static __device__ __forceinline__ void mma_set(FragC acc[2][4], const FragSet& s) {
  acc[0][0].v = __builtin_amdgcn_wmma_f32_16x16x32_bf16(false, s.a0, false, s.b0, (short)0, acc[0][0].v, false, false);
  acc[1][0].v = __builtin_amdgcn_wmma_f32_16x16x32_bf16(false, s.a1, false, s.b0, (short)0, acc[1][0].v, false, false);
  acc[0][1].v = __builtin_amdgcn_wmma_f32_16x16x32_bf16(false, s.a0, false, s.b1, (short)0, acc[0][1].v, false, false);
  acc[1][1].v = __builtin_amdgcn_wmma_f32_16x16x32_bf16(false, s.a1, false, s.b1, (short)0, acc[1][1].v, false, false);
  acc[0][2].v = __builtin_amdgcn_wmma_f32_16x16x32_bf16(false, s.a0, false, s.b2, (short)0, acc[0][2].v, false, false);
  acc[1][2].v = __builtin_amdgcn_wmma_f32_16x16x32_bf16(false, s.a1, false, s.b2, (short)0, acc[1][2].v, false, false);
  acc[0][3].v = __builtin_amdgcn_wmma_f32_16x16x32_bf16(false, s.a0, false, s.b3, (short)0, acc[0][3].v, false, false);
  acc[1][3].v = __builtin_amdgcn_wmma_f32_16x16x32_bf16(false, s.a1, false, s.b3, (short)0, acc[1][3].v, false, false);
}

__global__ __launch_bounds__(256) void gemm_bf16_wmma(
    const unsigned short* __restrict__ A,
    const unsigned short* __restrict__ Bt,
    float* __restrict__ C, int M, int N, int K) {
  const int wave = threadIdx.x >> 5;
  const int lane = threadIdx.x & 31;
  const int m0 = blockIdx.y * 256 + wave * 32;
  const int n0 = blockIdx.x * 64;

  FragC acc[2][4];
#pragma unroll
  for (int i = 0; i < 2; ++i)
#pragma unroll
    for (int t = 0; t < 4; ++t) { v8f z = {}; acc[i][t].v = z; }

  const unsigned short* A0  = A  + (size_t)m0 * K;
  const unsigned short* A1  = A  + (size_t)(m0 + 16) * K;
  const unsigned short* Br0 = Bt + (size_t)(n0 + 0)  * K;
  const unsigned short* Br1 = Bt + (size_t)(n0 + 16) * K;
  const unsigned short* Br2 = Bt + (size_t)(n0 + 32) * K;
  const unsigned short* Br3 = Bt + (size_t)(n0 + 48) * K;

  FragSet f0, f1;
  load_set(f0, A0, A1, Br0, Br1, Br2, Br3, 0, K);
  load_set(f1, A0, A1, Br0, Br1, Br2, Br3, 32, K);

  for (int k = 64; k < K; k += 64) {
    mma_set(acc, f0);                 // consume step k-64 (loads long done)
    load_set(f0, A0, A1, Br0, Br1, Br2, Br3, k, K);
    __builtin_prefetch(A0 + k + 64, 0, 1);
    __builtin_prefetch(Br0 + k + 64, 0, 1);
    mma_set(acc, f1);                 // consume step k-32, overlaps f0 loads
    load_set(f1, A0, A1, Br0, Br1, Br2, Br3, k + 32, K);
  }
  mma_set(acc, f0);
  mma_set(acc, f1);

  const int half = lane >> 4;
  const int nn   = lane & 15;
#pragma unroll
  for (int i = 0; i < 2; ++i)
#pragma unroll
    for (int t = 0; t < 4; ++t)
#pragma unroll
      for (int v = 0; v < 8; ++v) {
        int m = m0 + i * 16 + v + half * 8;
        C[(size_t)m * N + n0 + t * 16 + nn] = acc[i][t].f[v];
      }
}

// ---------------- flash attention (one wave per (b,h,qtile)) ----------------
__global__ __launch_bounds__(32) void attn_fwd_wmma(
    const unsigned short* __restrict__ Q,   // [B,H,T,Dh]  (roped)
    const unsigned short* __restrict__ Kr,  // [B,KV,T,Dh] (roped)
    const unsigned short* __restrict__ Vt,  // [B,KV,Dh,T]
    unsigned short* __restrict__ O) {       // [B,T,H*Dh]
  __shared__ unsigned short Pl[16 * 32];
  const int qt = blockIdx.x, h = blockIdx.y, b = blockIdx.z;
  const int hkv  = h / REP_;
  const int lane = threadIdx.x & 31;
  const int half = lane >> 4;
  const int nn   = lane & 15;

  const unsigned short* Qbase = Q + (((size_t)b * H_ + h) * T_ + qt * 16) * DH_;
  v16bf qf[4];
#pragma unroll
  for (int kk = 0; kk < 4; ++kk) qf[kk] = load_frag_16x32(Qbase + kk * 32, DH_);

  FragC o[8];
  float mrun[8], lrun[8];
#pragma unroll
  for (int nt = 0; nt < 8; ++nt) { v8f z = {}; o[nt].v = z; }
#pragma unroll
  for (int v = 0; v < 8; ++v) { mrun[v] = -1e30f; lrun[v] = 0.0f; }

  const float scale = 0.08838834764831845f;  // 1/sqrt(128)
  const int nPair = (qt * 16 + 15) / 32 + 1; // causal: key tiles of 32
  const unsigned short* Kb = Kr + ((size_t)b * KV_ + hkv) * (size_t)T_ * DH_;
  const unsigned short* Vb = Vt + ((size_t)b * KV_ + hkv) * (size_t)DH_ * T_;

  for (int j = 0; j < nPair; ++j) {
    const int key0 = j * 32;
    FragC s0, s1;
    { v8f z = {}; s0.v = z; s1.v = z; }
#pragma unroll
    for (int kk = 0; kk < 4; ++kk) {
      v16bf b0 = load_frag_16x32(Kb + (size_t)key0 * DH_ + kk * 32, DH_);
      s0.v = __builtin_amdgcn_wmma_f32_16x16x32_bf16(
          false, qf[kk], false, b0, (short)0, s0.v, false, false);
      v16bf b1 = load_frag_16x32(Kb + (size_t)(key0 + 16) * DH_ + kk * 32, DH_);
      s1.v = __builtin_amdgcn_wmma_f32_16x16x32_bf16(
          false, qf[kk], false, b1, (short)0, s1.v, false, false);
    }

    // Issue all V-fragment loads now: independent of softmax, so the VALU
    // softmax below hides their latency before the PV WMMAs.
    v16bf vfr[8];
#pragma unroll
    for (int nt = 0; nt < 8; ++nt)
      vfr[nt] = load_frag_16x32(Vb + (size_t)(nt * 16) * T_ + key0, T_);

    // scale + causal mask (per-lane selects; EXEC stays all-ones)
#pragma unroll
    for (int v = 0; v < 8; ++v) {
      const int qrow = qt * 16 + v + half * 8;
      float a0 = s0.f[v] * scale, a1 = s1.f[v] * scale;
      a0 = (key0 + nn      > qrow) ? -1e30f : a0;
      a1 = (key0 + 16 + nn > qrow) ? -1e30f : a1;
      s0.f[v] = a0; s1.f[v] = a1;
    }
    // online softmax (rows live in one VGPR across one 16-lane half)
#pragma unroll
    for (int v = 0; v < 8; ++v) {
      float rm = fmaxf(s0.f[v], s1.f[v]);
#pragma unroll
      for (int m = 1; m < 16; m <<= 1) rm = fmaxf(rm, __shfl_xor(rm, m, 32));
      float mn    = fmaxf(mrun[v], rm);
      float alpha = __expf(mrun[v] - mn);
      float p0 = __expf(s0.f[v] - mn);
      float p1 = __expf(s1.f[v] - mn);
      float rs = p0 + p1;
#pragma unroll
      for (int m = 1; m < 16; m <<= 1) rs += __shfl_xor(rs, m, 32);
      lrun[v] = lrun[v] * alpha + rs;
      mrun[v] = mn;
#pragma unroll
      for (int nt = 0; nt < 8; ++nt) o[nt].f[v] *= alpha;
      const int prow = v + half * 8;
      Pl[prow * 32 + nn]      = f2bf(p0);
      Pl[prow * 32 + 16 + nn] = f2bf(p1);
    }
    __syncthreads();                 // C-layout -> A-layout via LDS
    v16bf pf = load_frag_16x32(Pl, 32);
#pragma unroll
    for (int nt = 0; nt < 8; ++nt) {
      o[nt].v = __builtin_amdgcn_wmma_f32_16x16x32_bf16(
          false, pf, false, vfr[nt], (short)0, o[nt].v, false, false);
    }
    __syncthreads();
  }

#pragma unroll
  for (int v = 0; v < 8; ++v) {
    const int t = qt * 16 + v + half * 8;
    const float inv = 1.0f / lrun[v];
#pragma unroll
    for (int nt = 0; nt < 8; ++nt)
      O[(((size_t)b * T_ + t) * H_ + h) * DH_ + nt * 16 + nn] =
          f2bf(o[nt].f[v] * inv);
  }
}

// ---------------- host-side launcher ----------------
extern "C" void kernel_launch(void* const* d_in, const int* in_sizes, int n_in,
                              void* d_out, int out_size, void* d_ws, size_t ws_size,
                              hipStream_t stream) {
  const float* x  = (const float*)d_in[0];
  const float* Wq = (const float*)d_in[1];
  const float* Wk = (const float*)d_in[2];
  const float* Wv = (const float*)d_in[3];
  const float* Wo = (const float*)d_in[4];
  float* out = (float*)d_out;

  char* ws = (char*)d_ws;
  size_t off = 0;
  auto alloc = [&](size_t bytes) -> void* {
    void* p = ws + off;
    off += bytes;
    off = (off + 255) & ~(size_t)255;
    return p;
  };

  unsigned short* xb  = (unsigned short*)alloc((size_t)MTOK_ * D_ * 2);
  unsigned short* wqt = (unsigned short*)alloc((size_t)NQ_ * D_ * 2);
  unsigned short* wkt = (unsigned short*)alloc((size_t)NKV_ * D_ * 2);
  unsigned short* wvt = (unsigned short*)alloc((size_t)NKV_ * D_ * 2);
  unsigned short* wot = (unsigned short*)alloc((size_t)D_ * NQ_ * 2);
  float* qf = (float*)alloc((size_t)MTOK_ * NQ_ * 4);
  float* kf = (float*)alloc((size_t)MTOK_ * NKV_ * 4);
  float* vf = (float*)alloc((size_t)MTOK_ * NKV_ * 4);
  unsigned short* Qr = (unsigned short*)alloc((size_t)MTOK_ * NQ_ * 2);
  unsigned short* Kv = (unsigned short*)alloc((size_t)MTOK_ * NKV_ * 2);
  unsigned short* Vt = (unsigned short*)alloc((size_t)MTOK_ * NKV_ * 2);
  unsigned short* Ab = (unsigned short*)alloc((size_t)MTOK_ * NQ_ * 2);
  (void)ws_size; (void)in_sizes; (void)n_in; (void)out_size;

  int n;
  n = MTOK_ * D_;
  cvt_bf16_kernel<<<(n + 255) / 256, 256, 0, stream>>>(x, xb, n);
  n = D_ * NQ_;
  wt_cvt_kernel<<<(n + 255) / 256, 256, 0, stream>>>(Wq, wqt, D_, NQ_);
  n = D_ * NKV_;
  wt_cvt_kernel<<<(n + 255) / 256, 256, 0, stream>>>(Wk, wkt, D_, NKV_);
  wt_cvt_kernel<<<(n + 255) / 256, 256, 0, stream>>>(Wv, wvt, D_, NKV_);
  n = NQ_ * D_;
  wt_cvt_kernel<<<(n + 255) / 256, 256, 0, stream>>>(Wo, wot, NQ_, D_);

  // QKV projections (block tile 256M x 64N, wave tile 32x64)
  gemm_bf16_wmma<<<dim3(NQ_ / 64, MTOK_ / 256), 256, 0, stream>>>(
      xb, wqt, qf, MTOK_, NQ_, D_);
  gemm_bf16_wmma<<<dim3(NKV_ / 64, MTOK_ / 256), 256, 0, stream>>>(
      xb, wkt, kf, MTOK_, NKV_, D_);
  gemm_bf16_wmma<<<dim3(NKV_ / 64, MTOK_ / 256), 256, 0, stream>>>(
      xb, wvt, vf, MTOK_, NKV_, D_);

  // RoPE + layout conversions
  n = B_ * T_ * H_ * (DH_ / 2);
  rope_cvt_kernel<<<(n + 255) / 256, 256, 0, stream>>>(qf, Qr, H_);
  n = B_ * T_ * KV_ * (DH_ / 2);
  rope_cvt_kernel<<<(n + 255) / 256, 256, 0, stream>>>(kf, Kv, KV_);
  n = MTOK_ * NKV_;
  vtrans_cvt_kernel<<<(n + 255) / 256, 256, 0, stream>>>(vf, Vt);

  // Flash attention
  attn_fwd_wmma<<<dim3(T_ / 16, H_, B_), 32, 0, stream>>>(Qr, Kv, Vt, Ab);

  // Output projection -> d_out (f32)
  gemm_bf16_wmma<<<dim3(D_ / 64, MTOK_ / 256), 256, 0, stream>>>(
      Ab, wot, out, MTOK_, D_, D_);
}
